// InterpretableMultiHeadAttention_66468913873501
// MI455X (gfx1250) — matrix operand order repaired and checked by reading, not compiled, not run on hardware
//
#include <hip/hip_runtime.h>
#include <hip/hip_bf16.h>

// ---------------------------------------------------------------------------
// InterpretableMultiHeadAttention for MI455X (gfx1250), bf16 WMMA pipeline.
//
// Shapes: B=2, T=1024, D=1024, H=8 outer heads, 8 inner heads of d=128.
// outputs:   d_out[0 .. B*T*D)              fp32
// attention: d_out[B*T*D .. +B*H*T*T)       fp32
//
// Precision: all GEMMs in bf16 (v_wmma_f32_16x16x32_bf16, fp32 accum),
// softmax in fp32. HBM floor ~8us; bf16 WMMA keeps compute in the same
// order of magnitude, fp32 VALU would be ~50x slower.
//
// Workspace layout (assumed ws_size >= ~80 MB):
//   qp  bf16 [H][B*T][D]   33,554,432 B
//   kp  bf16 [H][B*T][D]   33,554,432 B
//   vp  bf16 [B*T][D]       4,194,304 B
//   mh  fp32 [B*T][D]       8,388,608 B   (mean over outer heads)
// ---------------------------------------------------------------------------

typedef __bf16 v16bf __attribute__((ext_vector_type(16)));
typedef float  v8f   __attribute__((ext_vector_type(8)));
typedef unsigned int   u32;
typedef unsigned short u16;
typedef u32   u32x4 __attribute__((ext_vector_type(4)));
typedef u32   u32x2 __attribute__((ext_vector_type(2)));
typedef float f32x4 __attribute__((ext_vector_type(4)));

#define DEV static __device__ __forceinline__

constexpr int Bb    = 2;
constexpr int T     = 1024;
constexpr int Dm    = 1024;
constexpr int H     = 8;
constexpr int HD    = 128;      // inner head dim
constexpr int Mrows = Bb * T;   // 2048

union AFrag { u32x4 u[2]; v16bf v; };

DEV u16 f2bf(float f) {                    // round-to-nearest-even fp32->bf16
  u32 x = __float_as_uint(f);
  x += 0x7FFFu + ((x >> 16) & 1u);
  return (u16)(x >> 16);
}
DEV u32 pack2(float a, float b) { return (u32)f2bf(a) | ((u32)f2bf(b) << 16); }

DEV v8f wmma_bf16(v16bf a, v16bf b, v8f c) {
  // D = A(16x32 bf16) * B(32x16 bf16) + C(16x16 f32)
  return __builtin_amdgcn_wmma_f32_16x16x32_bf16(false, a, false, b,
                                                 (short)0, c, false, false);
}

// Build a 16-bf16 fragment from two 16-byte chunks (A/B per-lane layouts are
// two disjoint 16B runs per lane; see CDNA5 ISA 7.12.2).
DEV v16bf load_frag(const void* p0, const void* p1) {
  AFrag f;
  f.u[0] = *(const u32x4*)p0;
  f.u[1] = *(const u32x4*)p1;
  return f.v;
}

// ---------------------------------------------------------------------------
// Generic tiled GEMM: C(MxN) = A(MxK) * W(NxK)^T, K=N=Dm, fp32 inputs
// converted to bf16 while staging into LDS. Block = 256 threads = 8 waves,
// 128x128 output tile; wave computes 64x32 (4x2 WMMA tiles).
// ---------------------------------------------------------------------------
template <bool BF16OUT>
DEV void gemm_128x128(const float* __restrict__ A, const float* __restrict__ W,
                      void* __restrict__ Cout) {
  __shared__ u16 sA[128 * 64];
  __shared__ u16 sW[128 * 64];

  const int tid  = threadIdx.x;
  const int wave = tid >> 5;
  const int lane = tid & 31;
  const int half = lane >> 4;          // which 16-lane half of the wave
  const int l16  = lane & 15;
  const int m0   = blockIdx.x * 128;
  const int n0   = blockIdx.y * 128;
  const int wm   = (wave >> 2) * 64;   // wave's M offset in tile
  const int wn   = (wave & 3) * 32;    // wave's N offset in tile

  v8f acc[4][2];
#pragma unroll
  for (int mt = 0; mt < 4; ++mt)
#pragma unroll
    for (int nt = 0; nt < 2; ++nt)
#pragma unroll
      for (int r = 0; r < 8; ++r) acc[mt][nt][r] = 0.0f;

  for (int k0 = 0; k0 < Dm; k0 += 64) {
    __syncthreads();
    // Stage A[128][64] and W[128][64] as bf16 into LDS (float4 reads).
#pragma unroll
    for (int jj = 0; jj < 8; ++jj) {
      const int e4  = tid + jj * 256;       // 2048 float4 groups per operand
      const int row = e4 >> 4;
      const int c4  = (e4 & 15) * 4;
      f32x4 ga = *(const f32x4*)&A[(size_t)(m0 + row) * Dm + k0 + c4];
      f32x4 gw = *(const f32x4*)&W[(size_t)(n0 + row) * Dm + k0 + c4];
      u32x2 pa, pw;
      pa.x = pack2(ga.x, ga.y); pa.y = pack2(ga.z, ga.w);
      pw.x = pack2(gw.x, gw.y); pw.y = pack2(gw.z, gw.w);
      *(u32x2*)&sA[row * 64 + c4] = pa;
      *(u32x2*)&sW[row * 64 + c4] = pw;
      if (k0 + 64 < Dm) {  // global_prefetch_b8 of next K slab
        __builtin_prefetch(&A[(size_t)(m0 + row) * Dm + k0 + 64 + c4], 0, 1);
        __builtin_prefetch(&W[(size_t)(n0 + row) * Dm + k0 + 64 + c4], 0, 1);
      }
    }
    __syncthreads();

#pragma unroll
    for (int kk = 0; kk < 64; kk += 32) {
      v16bf af[4], bf[2];
#pragma unroll
      for (int mt = 0; mt < 4; ++mt) {
        // A frag: lane half 0 -> K[0..16); half 1 -> K[16..32), contiguous
        const int base = (wm + mt * 16 + l16) * 64 + kk + half * 16;
        af[mt] = load_frag(&sA[base], &sA[base + 8]);
      }
#pragma unroll
      for (int nt = 0; nt < 2; ++nt) {
        // B frag: lane half 0 -> K[0..16); half 1 -> K[16..32), contiguous
        const int base = (wn + nt * 16 + l16) * 64 + kk + half * 16;
        bf[nt] = load_frag(&sW[base], &sW[base + 8]);
      }
#pragma unroll
      for (int mt = 0; mt < 4; ++mt)
#pragma unroll
        for (int nt = 0; nt < 2; ++nt)
          acc[mt][nt] = wmma_bf16(af[mt], bf[nt], acc[mt][nt]);
    }
  }

  // C/D layout: VGPR r -> row r (lanes 0-15) / row r+8 (lanes 16-31), col=l16
#pragma unroll
  for (int mt = 0; mt < 4; ++mt)
#pragma unroll
    for (int nt = 0; nt < 2; ++nt)
#pragma unroll
      for (int r = 0; r < 8; ++r) {
        const int row = m0 + wm + mt * 16 + r + half * 8;
        const int col = n0 + wn + nt * 16 + l16;
        const float v = acc[mt][nt][r];
        if (BF16OUT) ((u16*)Cout)[(size_t)row * Dm + col] = f2bf(v);
        else         ((float*)Cout)[(size_t)row * Dm + col] = v;
      }
}

// grid (16, 8, 17): z<8 -> Q proj per outer head, z<16 -> K proj, z==16 -> V
__global__ __launch_bounds__(256) void proj_gemm_kernel(
    const float* __restrict__ q, const float* __restrict__ k,
    const float* __restrict__ v, const float* __restrict__ Wq,
    const float* __restrict__ Wk, const float* __restrict__ Wv,
    u16* __restrict__ qp, u16* __restrict__ kp, u16* __restrict__ vp) {
  const int z = blockIdx.z;
  const float* A;
  const float* W;
  u16* C;
  if (z < 8)       { A = q; W = Wq + (size_t)z * Dm * Dm;      C = qp + (size_t)z * Mrows * Dm; }
  else if (z < 16) { A = k; W = Wk + (size_t)(z - 8) * Dm * Dm; C = kp + (size_t)(z - 8) * Mrows * Dm; }
  else             { A = v; W = Wv;                             C = vp; }
  gemm_128x128<true>(A, W, C);
}

__global__ __launch_bounds__(256) void out_gemm_kernel(
    const float* __restrict__ mh, const float* __restrict__ Wo,
    float* __restrict__ out) {
  gemm_128x128<false>(mh, Wo, out);
}

// ---------------------------------------------------------------------------
// Attention: grid (T/64, H, B), 128 threads (4 waves x 16 query rows).
// For each outer head i: pass 1 = online row max/sumexp over score tiles
// (WMMA QK^T); pass 2 = recompute scores, emit mean probabilities to d_out,
// repack P to bf16 A-fragments via per-wave LDS, accumulate P*V with WMMA
// against a V block transposed into LDS. Head sum over i -> *0.125 -> mh.
// ---------------------------------------------------------------------------
__global__ __launch_bounds__(128) void attn_kernel(
    const u16* __restrict__ qp, const u16* __restrict__ kp,
    const u16* __restrict__ vp, float* __restrict__ attn,
    float* __restrict__ meanhead) {
  __shared__ u16 ldsT[128 * 136];     // V chunk transposed: [d(128)][s(128)+pad]
  __shared__ u16 pbuf[4][16 * 32];    // per-wave P repack buffer (row-major)

  const int tid  = threadIdx.x;
  const int wave = tid >> 5;
  const int lane = tid & 31;
  const int half = lane >> 4;
  const int l16  = lane & 15;
  const int b    = blockIdx.z;
  const int h    = blockIdx.y;
  const int t0   = blockIdx.x * 64 + wave * 16;
  const float inv_scale = 0.0883883476483184f;  // 1/sqrt(128)

  v8f hacc[8];
#pragma unroll
  for (int dt = 0; dt < 8; ++dt)
#pragma unroll
    for (int r = 0; r < 8; ++r) hacc[dt][r] = 0.0f;

  for (int i = 0; i < H; ++i) {
    const size_t qkbase = ((size_t)i * Bb + b) * T;  // row base for qp/kp of head i

    // Cache Q fragments for all 4 k-steps over d (reused across all of s).
    v16bf qf[4];
    {
      const u16* qrow = qp + (qkbase + t0 + l16) * (size_t)Dm + h * HD;
#pragma unroll
      for (int dd = 0; dd < 4; ++dd) {
        const u16* p = qrow + dd * 32 + half * 16;
        qf[dd] = load_frag(p, p + 8);
      }
    }

    // ---- pass 1: per-row max and sum(exp) ----
    float m[8], l[8];
#pragma unroll
    for (int r = 0; r < 8; ++r) { m[r] = -1e30f; l[r] = 0.0f; }

    for (int s0 = 0; s0 < T; s0 += 16) {
      v8f c;
#pragma unroll
      for (int r = 0; r < 8; ++r) c[r] = 0.0f;
      const u16* krow = kp + (qkbase + s0 + l16) * (size_t)Dm + h * HD;
#pragma unroll
      for (int dd = 0; dd < 4; ++dd) {
        const u16* p = krow + dd * 32 + half * 16;
        c = wmma_bf16(qf[dd], load_frag(p, p + 8), c);
      }
#pragma unroll
      for (int r = 0; r < 8; ++r) {
        const float v  = c[r] * inv_scale;
        const float nm = fmaxf(m[r], v);
        l[r] = l[r] * __expf(m[r] - nm) + __expf(v - nm);
        m[r] = nm;
      }
    }
    // Merge the 16 lane-local partials of each tile half (rows r / r+8).
#pragma unroll
    for (int off = 1; off < 16; off <<= 1) {
#pragma unroll
      for (int r = 0; r < 8; ++r) {
        const float om = __shfl_xor(m[r], off, 32);
        const float ol = __shfl_xor(l[r], off, 32);
        const float nm = fmaxf(m[r], om);
        l[r] = l[r] * __expf(m[r] - nm) + ol * __expf(om - nm);
        m[r] = nm;
      }
    }
    float rl[8];
#pragma unroll
    for (int r = 0; r < 8; ++r) rl[r] = 1.0f / l[r];

    // ---- pass 2: probabilities out + P*V ----
    for (int sc = 0; sc < T; sc += 128) {
      __syncthreads();
      {  // stage V[sc..sc+128][h*128..+128] transposed into LDS (bf16)
        const u16* vrow = vp + ((size_t)b * T + sc + tid) * Dm + h * HD;
#pragma unroll 8
        for (int d = 0; d < HD; ++d) ldsT[d * 136 + tid] = vrow[d];
      }
      __syncthreads();

#pragma unroll
      for (int ssub = 0; ssub < 4; ++ssub) {
        const int s0 = sc + ssub * 32;
        v8f c0, c1;
#pragma unroll
        for (int r = 0; r < 8; ++r) { c0[r] = 0.0f; c1[r] = 0.0f; }
        const u16* krow0 = kp + (qkbase + s0 + l16) * (size_t)Dm + h * HD;
        const u16* krow1 = krow0 + (size_t)16 * Dm;
#pragma unroll
        for (int dd = 0; dd < 4; ++dd) {
          const u16* p0 = krow0 + dd * 32 + half * 16;
          const u16* p1 = krow1 + dd * 32 + half * 16;
          c0 = wmma_bf16(qf[dd], load_frag(p0, p0 + 8), c0);
          c1 = wmma_bf16(qf[dd], load_frag(p1, p1 + 8), c1);
        }

        u16* pb = pbuf[wave];
        float* arow = attn + ((size_t)b * H + h) * T * (size_t)T;
#pragma unroll
        for (int r = 0; r < 8; ++r) {
          const float pr0 = __expf(c0[r] * inv_scale - m[r]) * rl[r];
          const float pr1 = __expf(c1[r] * inv_scale - m[r]) * rl[r];
          const int trow  = t0 + r + half * 8;
          float* ap = arow + (size_t)trow * T + s0 + l16;
          if (i == 0) { ap[0]  = pr0 * 0.125f; ap[16]  = pr1 * 0.125f; }
          else        { ap[0] += pr0 * 0.125f; ap[16] += pr1 * 0.125f; }
          pb[(r + half * 8) * 32 + l16]      = f2bf(pr0);
          pb[(r + half * 8) * 32 + 16 + l16] = f2bf(pr1);
        }
        // wave-local store->load hazard on pbuf (no cross-wave sharing)
        asm volatile("s_wait_dscnt 0" ::: "memory");

        const u16* pp = pb + l16 * 32 + half * 16;   // P as A-fragment (K = s)
        v16bf pf = load_frag(pp, pp + 8);
#pragma unroll
        for (int dt = 0; dt < 8; ++dt) {
          const u16* vpp = &ldsT[(dt * 16 + l16) * 136 + ssub * 32 + half * 16];
          hacc[dt] = wmma_bf16(pf, load_frag(vpp, vpp + 8), hacc[dt]);
        }
      }
    }
  }

  // mean over outer heads -> fp32 workspace for the final projection
#pragma unroll
  for (int dt = 0; dt < 8; ++dt)
#pragma unroll
    for (int r = 0; r < 8; ++r) {
      const int trow = t0 + r + half * 8;
      const int col  = h * HD + dt * 16 + l16;
      meanhead[((size_t)b * T + trow) * Dm + col] = hacc[dt][r] * 0.125f;
    }
}

// ---------------------------------------------------------------------------
extern "C" void kernel_launch(void* const* d_in, const int* in_sizes, int n_in,
                              void* d_out, int out_size, void* d_ws,
                              size_t ws_size, hipStream_t stream) {
  (void)in_sizes; (void)n_in; (void)out_size; (void)ws_size;

  const float* q  = (const float*)d_in[0];
  const float* k  = (const float*)d_in[1];
  const float* v  = (const float*)d_in[2];
  const float* Wq = (const float*)d_in[3];
  const float* Wk = (const float*)d_in[4];
  const float* Wv = (const float*)d_in[5];
  const float* Wo = (const float*)d_in[6];

  float* out  = (float*)d_out;                       // (B,T,D)
  float* attn = out + (size_t)Bb * T * Dm;           // (B,H,T,T)

  char* ws = (char*)d_ws;
  u16*   qp = (u16*)ws;                                        // 33,554,432 B
  u16*   kp = (u16*)(ws + (size_t)33554432);                   // 33,554,432 B
  u16*   vp = (u16*)(ws + (size_t)67108864);                   //  4,194,304 B
  float* mh = (float*)(ws + (size_t)71303168);                 //  8,388,608 B

  // 1) 17 projection GEMMs (8x Q, 8x K, 1x V), bf16 outputs.
  proj_gemm_kernel<<<dim3(Mrows / 128, Dm / 128, 17), 256, 0, stream>>>(
      q, k, v, Wq, Wk, Wv, qp, kp, vp);

  // 2) attention + mean over outer heads (writes attn to d_out, mh to ws).
  attn_kernel<<<dim3(T / 64, H, Bb), 128, 0, stream>>>(qp, kp, vp, attn, mh);

  // 3) output projection: mean_head @ Wo^T -> fp32 outputs.
  out_gemm_kernel<<<dim3(Mrows / 128, Dm / 128, 1), 256, 0, stream>>>(mh, Wo, out);
}
